// HYPHEN_41472204210324
// MI455X (gfx1250) — compile-verified, matrix operand order checked by reading, not run on hardware
//
#include <hip/hip_runtime.h>
#include <hip/hip_bf16.h>

// ---------------------------------------------------------------------------
// HYPHEN (hyperbolic LSTM + attention) for MI455X / gfx1250.
// B=128, S=384, H=384, IN=384, NC=2.
// Recurrent GEMMs via v_wmma_f32_16x16x32_bf16; Mobius algebra via per-batch
// blocks with LDS tree reductions (wave32-safe, uniform control flow).
// ---------------------------------------------------------------------------

#define BB   128
#define SS   384
#define HH   384
#define G4   1536          // 4*H
#define EPSF 1e-15f
#define MAXN (1.0f - 1e-3f)

typedef __attribute__((ext_vector_type(16))) __bf16 v16bf;
typedef __attribute__((ext_vector_type(8)))  float  v8f;

union FragAB { v16bf v; uint4 q[2]; };

// ---------------------------------------------------------------------------
// helpers
// ---------------------------------------------------------------------------
__device__ __forceinline__ float artanh_c(float x) {
  x = fminf(fmaxf(x, -1.0f + 1e-7f), 1.0f - 1e-7f);
  return atanhf(x);
}

__device__ __forceinline__ unsigned short f2bf(float f) {
  unsigned int u = __float_as_uint(f);
  unsigned int r = u + 0x7FFFu + ((u >> 16) & 1u);   // round-to-nearest-even
  return (unsigned short)(r >> 16);
}

// block-wide sum over 384 threads (pads to 512 in LDS)
__device__ __forceinline__ float block_sum(float v, float* red) {
  int t = threadIdx.x;
  red[t] = v;
  if (t < 128) red[t + 384] = 0.0f;
  __syncthreads();
  for (int off = 256; off > 0; off >>= 1) {
    if (t < off) red[t] += red[t + off];
    __syncthreads();
  }
  float r = red[0];
  __syncthreads();
  return r;
}

__device__ __forceinline__ float block_max(float v, float* red) {
  int t = threadIdx.x;
  red[t] = v;
  if (t < 128) red[t + 384] = -3.4e38f;
  __syncthreads();
  for (int off = 256; off > 0; off >>= 1) {
    if (t < off) red[t] = fmaxf(red[t], red[t + off]);
    __syncthreads();
  }
  float r = red[0];
  __syncthreads();
  return r;
}

// mobius_matvec elementwise: res = tanh(mxn/srcn * artanh(srcn)) * mx / mxn
__device__ __forceinline__ float mob_mv(float mx, float mxn, float srcn) {
  float r = tanhf(mxn / srcn * artanh_c(srcn)) * mx / mxn;
  return (mxn > 1e-14f) ? r : 0.0f;
}

// mobius_pw elementwise given wx element and norms
__device__ __forceinline__ float mob_pw(float wx, float wxn, float xn) {
  float r = tanhf(wxn / xn * artanh_c(xn)) * wx / wxn;
  return (wxn > 1e-14f) ? r : 0.0f;
}

// mobius_add elementwise given scalar invariants
__device__ __forceinline__ float mob_add(float x, float y, float x2, float y2, float xy) {
  float num = (1.0f + 2.0f * xy + y2) * x + (1.0f - x2) * y;
  float den = fmaxf(1.0f + 2.0f * xy + x2 * y2, EPSF);
  return num / den;
}

// ---------------------------------------------------------------------------
// conversion / init kernels
// ---------------------------------------------------------------------------
__global__ void hyp_conv_x(const float* __restrict__ x, unsigned short* __restrict__ xb,
                           float* __restrict__ xnorm2) {
  __shared__ float red[512];
  size_t r = blockIdx.x;                      // row over B*S
  int t = threadIdx.x;                        // 0..383
  float v = x[r * HH + t];
  xb[r * HH + t] = f2bf(v);
  float s2 = block_sum(v * v, red);
  if (t == 0) xnorm2[r] = s2;
}

// Wb[g*H+n][k] = W_all[n][g*H+k]   (so B operand is [N=1536, K=384] row-major)
__global__ void hyp_conv_wb(const float* __restrict__ W_all, unsigned short* __restrict__ Wb) {
  int idx = blockIdx.x * blockDim.x + threadIdx.x;
  if (idx >= G4 * HH) return;
  int row = idx / HH;          // g*H + n
  int k   = idx % HH;
  int gg  = row / HH;
  int n   = row % HH;
  Wb[idx] = f2bf(W_all[n * G4 + gg * HH + k]);
}

__global__ void hyp_conv_bf(const float* __restrict__ src, unsigned short* __restrict__ dst, int n) {
  int i = blockIdx.x * blockDim.x + threadIdx.x;
  if (i < n) dst[i] = f2bf(src[i]);
}

__global__ void hyp_zero_state(float* h, float* cc, unsigned short* hb, unsigned short* ccb) {
  int i = blockIdx.x * blockDim.x + threadIdx.x;
  if (i < BB * HH) { h[i] = 0.0f; cc[i] = 0.0f; hb[i] = 0; ccb[i] = 0; }
}

// ---------------------------------------------------------------------------
// per-step fused GEMM:  Wh = h@Wb^T-layout, Ux = x_s@U^T, Wdc = cc@Wd^T
// 216 blocks x 256 threads (8 waves); each wave -> one 16x16 WMMA tile.
// ---------------------------------------------------------------------------
__global__ void hyp_step_gemm(const unsigned short* __restrict__ hb,
                              const unsigned short* __restrict__ ccb,
                              const unsigned short* __restrict__ xb,
                              const unsigned short* __restrict__ Wb,
                              const unsigned short* __restrict__ Ub,
                              const unsigned short* __restrict__ Wdb,
                              float* __restrict__ Wh, float* __restrict__ Ux,
                              float* __restrict__ Wdc, int s) {
  int bid = blockIdx.x;
  const unsigned short* A; const unsigned short* Bm; float* O;
  int N; long astride; int rb, cbl;
  if (bid < 96) {                 // job0: h @ W   -> [128,1536]
    int j = bid; rb = j / 12; cbl = j % 12;
    A = hb; astride = HH; Bm = Wb; O = Wh; N = G4;
  } else if (bid < 192) {         // job1: x_s @ U -> [128,1536]
    int j = bid - 96; rb = j / 12; cbl = j % 12;
    A = xb + (long)s * HH; astride = (long)SS * HH; Bm = Ub; O = Ux; N = G4;
  } else {                        // job2: cc @ Wd -> [128,384]
    int j = bid - 192; rb = j / 3; cbl = j % 3;
    A = ccb; astride = HH; Bm = Wdb; O = Wdc; N = HH;
  }

  int wave = threadIdx.x >> 5;
  int lane = threadIdx.x & 31;
  int lo = lane & 15;
  int hi = lane >> 4;
  int m0 = rb * 16;
  int n0 = cbl * 128 + wave * 16;

  v8f acc = {0.f, 0.f, 0.f, 0.f, 0.f, 0.f, 0.f, 0.f};
  const unsigned short* Arow = A  + (long)(m0 + lo) * astride;
  const unsigned short* Brow = Bm + (long)(n0 + lo) * HH;

  for (int kc = 0; kc < HH; kc += 32) {
    FragAB a, b;
    int ka = kc + hi * 8;
    a.q[0] = *(const uint4*)(Arow + ka);
    a.q[1] = *(const uint4*)(Arow + ka + 16);
    b.q[0] = *(const uint4*)(Brow + ka);
    b.q[1] = *(const uint4*)(Brow + ka + 16);
    acc = __builtin_amdgcn_wmma_f32_16x16x32_bf16(
        false, a.v, false, b.v, (short)0, acc, false, false);
  }
#pragma unroll
  for (int r = 0; r < 8; r++) {
    O[(long)(m0 + r + hi * 8) * N + (n0 + lo)] = acc[r];
  }
}

// ---------------------------------------------------------------------------
// per-step hyperbolic update: one block per batch row b, 384 threads (t = dim)
// ---------------------------------------------------------------------------
__global__ void hyp_step_update(const float* __restrict__ Wh, const float* __restrict__ Ux,
                                const float* __restrict__ Wdc, const float* __restrict__ xnorm2,
                                const float* __restrict__ ts,
                                float* __restrict__ h, float* __restrict__ cc,
                                unsigned short* __restrict__ hb, unsigned short* __restrict__ ccb,
                                float* __restrict__ context, int s) {
  __shared__ float red[512];
  int b = blockIdx.x;
  int t = threadIdx.x;

  float hv = h[b * HH + t];
  float cv = cc[b * HH + t];
  float hn2 = block_sum(hv * hv, red);
  float cn2 = block_sum(cv * cv, red);
  float hn = sqrtf(fmaxf(hn2, EPSF));
  float cn = sqrtf(fmaxf(cn2, EPSF));
  float xn = sqrtf(fmaxf(xnorm2[(size_t)b * SS + s], EPSF));

  // ---- cell-decay path: c_s1 = expmap0(tanh(logmap0(mobius_matvec(W_d, cc))))
  float wdcv = Wdc[b * HH + t];
  float wdcn = sqrtf(fmaxf(block_sum(wdcv * wdcv, red), EPSF));
  float mcv  = mob_mv(wdcv, wdcn, cn);
  float mcn  = sqrtf(fmaxf(block_sum(mcv * mcv, red), EPSF));
  float uu   = artanh_c(mcn) / mcn * mcv;          // logmap0
  float vv   = tanhf(uu);
  float vvn  = sqrtf(fmaxf(block_sum(vv * vv, red), EPSF));
  float c1   = tanhf(vvn) / vvn * vv;              // expmap0
  float c1n2 = block_sum(c1 * c1, red);

  // c_s2 = mobius_pw(c_s1, broadcast(t_s))
  float tsv = ts[(size_t)b * SS + s];
  float tn  = sqrtf(fmaxf(block_sum(tsv * tsv, red), EPSF));
  float wxv = c1 * tsv;
  float wxn = sqrtf(fmaxf(block_sum(wxv * wxv, red), EPSF));
  float c2v = mob_pw(wxv, wxn, tn);

  // c_adj = mobius_add(mobius_add(-c1, cc), c_s2)
  float xy1 = block_sum(-c1 * cv, red);
  float t1  = mob_add(-c1, cv, c1n2, cn2, xy1);
  float t12 = block_sum(t1 * t1, red);
  float c22 = block_sum(c2v * c2v, red);
  float xy2 = block_sum(t1 * c2v, red);
  float cadj = mob_add(t1, c2v, t12, c22, xy2);

  // ---- gates f, i, o, ct
  float gates[4];
#pragma unroll 1
  for (int g = 0; g < 4; g++) {
    float whv = Wh[(size_t)b * G4 + g * HH + t];
    float uxv = Ux[(size_t)b * G4 + g * HH + t];
    float whn = sqrtf(fmaxf(block_sum(whv * whv, red), EPSF));
    float uxn = sqrtf(fmaxf(block_sum(uxv * uxv, red), EPSF));
    float am  = mob_mv(whv, whn, hn);
    float bm  = mob_mv(uxv, uxn, xn);
    float x2  = block_sum(am * am, red);
    float y2  = block_sum(bm * bm, red);
    float xy  = block_sum(am * bm, red);
    float ad  = mob_add(am, bm, x2, y2, xy);
    float adn = sqrtf(fmaxf(block_sum(ad * ad, red), EPSF));
    float lg  = artanh_c(adn) / adn * ad;          // logmap0
    gates[g]  = 1.0f / (1.0f + expf(-lg));
  }

  // cc_n = mobius_add(mobius_pw(i, ct), mobius_pw(f, c_adj))
  float ctn  = sqrtf(fmaxf(block_sum(gates[3] * gates[3], red), EPSF));
  float p1wx = gates[1] * gates[3];
  float p1wn = sqrtf(fmaxf(block_sum(p1wx * p1wx, red), EPSF));
  float p1   = mob_pw(p1wx, p1wn, ctn);
  float cadjn = sqrtf(fmaxf(block_sum(cadj * cadj, red), EPSF));
  float p2wx = gates[0] * cadj;
  float p2wn = sqrtf(fmaxf(block_sum(p2wx * p2wx, red), EPSF));
  float p2   = mob_pw(p2wx, p2wn, cadjn);
  float p12a = block_sum(p1 * p1, red);
  float p22a = block_sum(p2 * p2, red);
  float pxy  = block_sum(p1 * p2, red);
  float ccn  = mob_add(p1, p2, p12a, p22a, pxy);

  // h_n = mobius_pw(o, expmap0(tanh(cc_n)))
  float th  = tanhf(ccn);
  float thn = sqrtf(fmaxf(block_sum(th * th, red), EPSF));
  float ev  = tanhf(thn) / thn * th;               // expmap0
  float evn = sqrtf(fmaxf(block_sum(ev * ev, red), EPSF));
  float hwx = gates[2] * ev;
  float hwn = sqrtf(fmaxf(block_sum(hwx * hwx, red), EPSF));
  float hnv = mob_pw(hwx, hwn, evn);

  context[((size_t)b * SS + s) * HH + t] = gates[2];  // o
  h[b * HH + t] = hnv;   cc[b * HH + t] = ccn;
  hb[b * HH + t] = f2bf(hnv);
  ccb[b * HH + t] = f2bf(ccn);
}

// ---------------------------------------------------------------------------
// query = h @ Win.T
// ---------------------------------------------------------------------------
__global__ void hyp_query(const float* __restrict__ h, const float* __restrict__ Win,
                          float* __restrict__ q) {
  int b = blockIdx.x, t = threadIdx.x;
  const float* hr = h + b * HH;
  float acc = 0.0f;
  for (int k = 0; k < HH; k++) acc += hr[k] * Win[t * HH + k];
  q[b * HH + t] = acc;
}

// ---------------------------------------------------------------------------
// hyperbolic attention: one block per batch row, 384 threads
// ---------------------------------------------------------------------------
__global__ void hyp_attn(const float* __restrict__ q, const float* __restrict__ context,
                         const float* __restrict__ dt, const float* __restrict__ ae,
                         const float* __restrict__ ab, float* __restrict__ mixws,
                         float* __restrict__ combined) {
  __shared__ float red[512];
  __shared__ float qs[384], sc[384], bts[384], lam[384];
  int b = blockIdx.x, t = threadIdx.x;
  const float* ctx_b = context + (size_t)b * SS * HH;

  qs[t] = q[b * HH + t];
  __syncthreads();

  // ---- scores + softmax over s (t <-> s)
  const float* crow = ctx_b + (size_t)t * HH;
  float sco = 0.0f;
  for (int k = 0; k < HH; k++) sco += qs[k] * crow[k];
  float mx = block_max(sco, red);
  float ex = expf(sco - mx);
  float sm = block_sum(ex, red);
  float aw = ex / fmaxf(sm, EPSF);

  // aw = project(expmap0(aw))
  float awn = sqrtf(fmaxf(block_sum(aw * aw, red), EPSF));
  float awe = tanhf(awn) / awn * aw;
  float awen = sqrtf(fmaxf(block_sum(awe * awe, red), EPSF));
  float pfa0 = (awen > MAXN) ? MAXN / awen : 1.0f;
  sc[t] = awe * pfa0;
  __syncthreads();

  // ---- bt = project(expmap0(exp(-ab*dt)))   (t <-> s)
  float btr = expf(-ab[b] * dt[(size_t)b * SS + t]);
  float btn = sqrtf(fmaxf(block_sum(btr * btr, red), EPSF));
  float bte = tanhf(btn) / btn * btr;
  float bten = sqrtf(fmaxf(block_sum(bte * bte, red), EPSF));
  float pfb0 = (bten > MAXN) ? MAXN / bten : 1.0f;
  bts[t] = bte * pfb0;
  __syncthreads();
  float btfn = sqrtf(fmaxf(block_sum(bts[t] * bts[t], red), EPSF));

  // ---- per-hidden-row ops (t <-> h);  norms over s are thread-local loops
  float aev = ae[b];
  float xn2 = 0.0f, wxn2 = 0.0f;
  for (int s2 = 0; s2 < SS; s2++) {
    float xv = ctx_b[(size_t)s2 * HH + t];
    float wv = sc[s2] * xv;
    xn2 += xv * xv;  wxn2 += wv * wv;
  }
  float xn  = sqrtf(fmaxf(xn2, EPSF));
  float wxn = sqrtf(fmaxf(wxn2, EPSF));
  float msc = (wxn > 1e-14f) ? tanhf(wxn / xn * artanh_c(xn)) / wxn : 0.0f;
  float mrn = msc * wxn;                                   // ||mix row||
  float pfm = (mrn > MAXN) ? MAXN / mrn : 1.0f;            // project
  msc *= pfm;  mrn *= pfm;

  // tmp_a = project(mobius_pw(ae, mix))   (scalar weight)
  float wxa = fabsf(aev) * mrn;
  float asc = (wxa > 1e-14f) ? tanhf(wxa / fmaxf(mrn, 1e-20f) * artanh_c(mrn)) / wxa : 0.0f;
  float arn = asc * wxa;
  float pfa = (arn > MAXN) ? MAXN / arn : 1.0f;
  asc *= pfa;

  // write mix row; accumulate || tmp_a * bt || for second pw
  float* mrow = mixws + ((size_t)b * HH + t) * SS;
  float wxb2 = 0.0f;
  for (int s2 = 0; s2 < SS; s2++) {
    float xv = ctx_b[(size_t)s2 * HH + t];
    float mv = msc * sc[s2] * xv;
    mrow[s2] = mv;
    float tb = asc * aev * mv * bts[s2];
    wxb2 += tb * tb;
  }
  float wxnb = sqrtf(fmaxf(wxb2, EPSF));
  float bsc = (wxnb > 1e-14f) ? tanhf(wxnb / btfn * artanh_c(btfn)) / wxnb : 0.0f;
  float brn = bsc * wxnb;
  float pfb = (brn > MAXN) ? MAXN / brn : 1.0f;
  bsc *= pfb;

  // mobius_add(mix, relu(tmp_b)) invariants
  float y2 = 0.0f, xy = 0.0f;
  for (int s2 = 0; s2 < SS; s2++) {
    float mv = mrow[s2];
    float rv = fmaxf(bsc * asc * aev * mv * bts[s2], 0.0f);
    y2 += rv * rv;  xy += mv * rv;
  }
  float x2m = mrn * mrn;
  float nden = fmaxf(1.0f + 2.0f * xy + x2m * y2, EPSF);
  float cfa = (1.0f + 2.0f * xy + y2) / nden;
  float cfb = (1.0f - x2m) / nden;
  float nm2 = 0.0f;
  for (int s2 = 0; s2 < SS; s2++) {
    float mv = mrow[s2];
    float rv = fmaxf(bsc * asc * aev * mv * bts[s2], 0.0f);
    float nv = cfa * mv + cfb * rv;
    mrow[s2] = nv;
    nm2 += nv * nv;
  }
  float nmn = sqrtf(fmaxf(nm2, EPSF));
  float pf3 = (nmn > MAXN) ? MAXN / nmn : 1.0f;
  for (int s2 = 0; s2 < SS; s2++) mrow[s2] *= pf3;
  nm2 *= pf3 * pf3;
  lam[t] = 2.0f / fmaxf(1.0f - nm2, EPSF);
  __syncthreads();

  // ---- weighted midpoint over h; then logmap0 over s (t <-> s)
  float den = fmaxf(block_sum(lam[t] - 1.0f, red), 1e-10f);
  float nom = 0.0f;
  for (int h2 = 0; h2 < HH; h2++)
    nom += lam[h2] * mixws[((size_t)b * HH + h2) * SS + t];
  float v  = nom / den;
  float vn = sqrtf(fmaxf(block_sum(v * v, red), EPSF));
  float fin = tanhf(0.5f * artanh_c(vn)) / vn * v;        // mobius_scalar_mul(0.5)
  float fn  = sqrtf(fmaxf(block_sum(fin * fin, red), EPSF));
  float mlog = artanh_c(fn) / fn * fin;                   // logmap0

  combined[(size_t)b * (2 * HH) + t]      = mlog;
  combined[(size_t)b * (2 * HH) + HH + t] = qs[t];
}

// ---------------------------------------------------------------------------
// head: tanh(combined @ Wout.T) -> relu(@W1.T+b1) -> @W2.T+b2
// ---------------------------------------------------------------------------
__global__ void hyp_head(const float* __restrict__ combined, const float* __restrict__ Wout,
                         const float* __restrict__ W1, const float* __restrict__ b1,
                         const float* __restrict__ W2, const float* __restrict__ b2,
                         float* __restrict__ out) {
  __shared__ float a1[384], a2[384];
  int b = blockIdx.x, t = threadIdx.x;
  const float* cb = combined + (size_t)b * (2 * HH);
  float acc = 0.0f;
  for (int k = 0; k < 2 * HH; k++) acc += cb[k] * Wout[t * (2 * HH) + k];
  a1[t] = tanhf(acc);
  __syncthreads();
  float acc2 = 0.0f;
  for (int k = 0; k < HH; k++) acc2 += a1[k] * W1[t * HH + k];
  a2[t] = fmaxf(acc2 + b1[t], 0.0f);
  __syncthreads();
  if (t < 2) {
    float o = 0.0f;
    for (int k = 0; k < HH; k++) o += a2[k] * W2[t * HH + k];
    out[b * 2 + t] = o + b2[t];
  }
}

// ---------------------------------------------------------------------------
extern "C" void kernel_launch(void* const* d_in, const int* in_sizes, int n_in,
                              void* d_out, int out_size, void* d_ws, size_t ws_size,
                              hipStream_t stream) {
  const float* x       = (const float*)d_in[0];   // [B,S,IN]
  const float* tstamp  = (const float*)d_in[1];   // [B,S]
  const float* delta_t = (const float*)d_in[2];   // [B,1,S]
  const float* W_all   = (const float*)d_in[3];   // [H,4H]
  const float* U_all   = (const float*)d_in[4];   // [4H,IN]
  const float* W_d     = (const float*)d_in[5];   // [H,H]
  const float* Win     = (const float*)d_in[6];   // [H,H]
  const float* Wout    = (const float*)d_in[7];   // [H,2H]
  const float* ae      = (const float*)d_in[8];   // [B,1,1]
  const float* ab      = (const float*)d_in[9];   // [B,1,1]
  const float* W1      = (const float*)d_in[10];  // [H,H]
  const float* b1      = (const float*)d_in[11];  // [H]
  const float* W2      = (const float*)d_in[12];  // [NC,H]
  const float* b2      = (const float*)d_in[13];  // [NC]
  float* out = (float*)d_out;

  // workspace carve-out
  char* p = (char*)d_ws;
  auto take = [&](size_t bytes) -> void* {
    void* r = (void*)p;
    p += (bytes + 255) & ~(size_t)255;
    return r;
  };
  unsigned short* xb  = (unsigned short*)take((size_t)BB * SS * HH * 2);
  unsigned short* Wb  = (unsigned short*)take((size_t)G4 * HH * 2);
  unsigned short* Ub  = (unsigned short*)take((size_t)G4 * HH * 2);
  unsigned short* Wdb = (unsigned short*)take((size_t)HH * HH * 2);
  float* xnorm2 = (float*)take((size_t)BB * SS * 4);
  float* hbuf   = (float*)take((size_t)BB * HH * 4);
  float* ccbuf  = (float*)take((size_t)BB * HH * 4);
  unsigned short* hbf  = (unsigned short*)take((size_t)BB * HH * 2);
  unsigned short* ccbf = (unsigned short*)take((size_t)BB * HH * 2);
  float* Wh   = (float*)take((size_t)BB * G4 * 4);
  float* Ux   = (float*)take((size_t)BB * G4 * 4);
  float* Wdc  = (float*)take((size_t)BB * HH * 4);
  float* context = (float*)take((size_t)BB * SS * HH * 4);
  float* q       = (float*)take((size_t)BB * HH * 4);
  float* mixws   = (float*)take((size_t)BB * HH * SS * 4);
  float* combined = (float*)take((size_t)BB * 2 * HH * 4);

  // ---- precompute: bf16 conversions, x norms, state init
  hipLaunchKernelGGL(hyp_conv_x, dim3(BB * SS), dim3(HH), 0, stream, x, xb, xnorm2);
  hipLaunchKernelGGL(hyp_conv_wb, dim3((G4 * HH + 255) / 256), dim3(256), 0, stream, W_all, Wb);
  hipLaunchKernelGGL(hyp_conv_bf, dim3((G4 * HH + 255) / 256), dim3(256), 0, stream, U_all, Ub, G4 * HH);
  hipLaunchKernelGGL(hyp_conv_bf, dim3((HH * HH + 255) / 256), dim3(256), 0, stream, W_d, Wdb, HH * HH);
  hipLaunchKernelGGL(hyp_zero_state, dim3((BB * HH + 255) / 256), dim3(256), 0, stream,
                     hbuf, ccbuf, hbf, ccbf);

  // ---- sequential hyperbolic LSTM scan
  for (int s = 0; s < SS; s++) {
    hipLaunchKernelGGL(hyp_step_gemm, dim3(216), dim3(256), 0, stream,
                       hbf, ccbf, xb, Wb, Ub, Wdb, Wh, Ux, Wdc, s);
    hipLaunchKernelGGL(hyp_step_update, dim3(BB), dim3(HH), 0, stream,
                       Wh, Ux, Wdc, xnorm2, tstamp, hbuf, ccbuf, hbf, ccbf, context, s);
  }

  // ---- attention + head
  hipLaunchKernelGGL(hyp_query, dim3(BB), dim3(HH), 0, stream, hbuf, Win, q);
  hipLaunchKernelGGL(hyp_attn, dim3(BB), dim3(HH), 0, stream,
                     q, context, delta_t, ae, ab, mixws, combined);
  hipLaunchKernelGGL(hyp_head, dim3(BB), dim3(HH), 0, stream,
                     combined, Wout, W1, b1, W2, b2, out);
}
